// GQA_83786222010912
// MI455X (gfx1250) — matrix operand order repaired and checked by reading, compiled, not run on hardware
//
#include <hip/hip_runtime.h>
#include <hip/hip_bf16.h>

typedef __attribute__((ext_vector_type(16))) _Float16 v16h;
typedef __attribute__((ext_vector_type(8)))  _Float16 v8h;
typedef __attribute__((ext_vector_type(4)))  _Float16 v4h;
typedef __attribute__((ext_vector_type(8)))  float    v8f;
typedef __attribute__((ext_vector_type(4)))  float    v4f;
typedef __attribute__((ext_vector_type(4)))  int      v4i;

// ---------------- constants (match reference) ----------------
static constexpr int Bn = 2;
static constexpr int S  = 2048;
static constexpr int E  = 1024;
static constexpr int Hq = 16;
static constexpr int Hk = 4;
static constexpr int D  = 64;
static constexpr int RS = Bn * S;     // 4096 rows (B*S)
static constexpr int QD = Hq * D;     // 1024
static constexpr int KD = Hk * D;     // 256

__device__ __forceinline__ v8f wmma16x16x32(v16h a, v16h b, v8f c) {
  return __builtin_amdgcn_wmma_f32_16x16x32_f16(false, a, false, b, (short)0, c,
                                                false, false);
}

union HFrag {
  v16h v;
  v8h  h[2];
};

// ---- A fragment (16x32 MxK) from row-major f16 [., lda]
// lanes 0-15 (M=lane): K=0..15; lanes 16-31: K=16..31
__device__ __forceinline__ v16h load_a_frag(const _Float16* __restrict__ A, int lda,
                                            int row0, int k0, int lane) {
  int m = lane & 15, h = lane >> 4;
  const _Float16* p = A + (size_t)(row0 + m) * lda + k0;
  HFrag f;
  f.h[0] = *(const v8h*)(p + h * 16);
  f.h[1] = *(const v8h*)(p + h * 16 + 8);
  return f.v;
}

// ---- B fragment (32x16 KxN), matrix stored K-contiguous: B[k,n] = Bt[n,k]
// lanes 0-15 (N=lane): K=0..15; lanes 16-31: K=16..31  (b[i] <-> K=kb+i)
__device__ __forceinline__ v16h load_bT_frag(const _Float16* __restrict__ Bt, int ldk,
                                             int nrow0, int k0, int lane) {
  int n  = lane & 15;
  int kb = (lane >> 4) * 16;
  return *(const v16h*)(Bt + (size_t)(nrow0 + n) * ldk + k0 + kb);
}

// ---------------- async global -> LDS staging (CDNA5 path) ----------------
#if __has_builtin(__builtin_amdgcn_global_load_async_to_lds_b128)
#define ASYNC_LDS 1
typedef v4i __attribute__((address_space(1))) gv4i;
typedef v4i __attribute__((address_space(3))) lv4i;
#endif

__device__ __forceinline__ void stage_b128(_Float16* ldst, const _Float16* gsrc) {
#ifdef ASYNC_LDS
  gv4i* g = (gv4i*)const_cast<_Float16*>(gsrc);
  lv4i* l = (lv4i*)ldst;
  __builtin_amdgcn_global_load_async_to_lds_b128(g, l, 0, 0);
#else
  *(v8h*)ldst = *(const v8h*)gsrc;
#endif
}

__device__ __forceinline__ void stage_wait_barrier() {
#ifdef ASYNC_LDS
#if __has_builtin(__builtin_amdgcn_s_wait_asynccnt)
  __builtin_amdgcn_s_wait_asynccnt(0);
#else
  asm volatile("s_wait_asynccnt 0x0" ::: "memory");
#endif
#endif
  __syncthreads();
}

// ================= operand staging kernels =================
__global__ void cvt_f32_f16(const float* __restrict__ in, _Float16* __restrict__ out,
                            int n4) {
  int i = blockIdx.x * blockDim.x + threadIdx.x;
  if (i >= n4) return;
  v4f x = ((const v4f*)in)[i];
  v4h y;
#pragma unroll
  for (int j = 0; j < 4; ++j) y[j] = (_Float16)x[j];
  ((v4h*)out)[i] = y;
}

// Wt[n*K + k] = (f16) W[k*N + n]
__global__ void transpose_cvt(const float* __restrict__ W, _Float16* __restrict__ Wt,
                              int K, int N) {
  int idx = blockIdx.x * blockDim.x + threadIdx.x;
  if (idx >= K * N) return;
  int k = idx / N, n = idx - k * N;
  Wt[(size_t)n * K + k] = (_Float16)W[idx];
}

// ============ GEMM: C[M,N] = A[M,K](f16,rowmaj) @ Bt[N,K](f16)^T ============
// block 256 = 8 waves in 2(M)x4(N); wave tile 16x64 (4 wmma / K-step).
template <typename OT, bool TRANS_OUT>
__global__ void __launch_bounds__(256) gemm_tn(const _Float16* __restrict__ A,
                                               const _Float16* __restrict__ Bt,
                                               OT* __restrict__ C, int M, int N,
                                               int K) {
  int lane = threadIdx.x & 31;
  int wave = threadIdx.x >> 5;
  int row0 = blockIdx.y * 32 + (wave >> 2) * 16;
  int col0 = blockIdx.x * 256 + (wave & 3) * 64;

  v8f acc[4] = {};
  for (int k0 = 0; k0 < K; k0 += 32) {
    v16h a = load_a_frag(A, K, row0, k0, lane);
#pragma unroll
    for (int j = 0; j < 4; ++j) {
      v16h b = load_bT_frag(Bt, K, col0 + j * 16, k0, lane);
      acc[j] = wmma16x16x32(a, b, acc[j]);
    }
  }
  int n = lane & 15, mb = (lane >> 4) * 8;
#pragma unroll
  for (int j = 0; j < 4; ++j) {
#pragma unroll
    for (int r = 0; r < 8; ++r) {
      int row = row0 + mb + r, col = col0 + j * 16 + n;
      if (TRANS_OUT)
        C[(size_t)col * M + row] = (OT)acc[j][r];
      else
        C[(size_t)row * N + col] = (OT)acc[j][r];
    }
  }
}

// ================= RoPE (in-place on f16 buffer [RS, nh*64]) =================
__global__ void rope_f16(_Float16* __restrict__ buf, const int* __restrict__ pos,
                         int nh, int total) {
  int idx = blockIdx.x * blockDim.x + threadIdx.x;
  if (idx >= total) return;
  int i  = idx & 31;
  int h  = (idx >> 5) % nh;
  int rs = idx / (32 * nh);
  float p   = (float)pos[rs];
  float ang = p * __expf(-(float)i * 0.28782313662425572f);  // ln(10000)/32
  float sv  = __sinf(ang), cv = __cosf(ang);
  _Float16* ptr = buf + (size_t)rs * (nh * 64) + h * 64 + i;
  float x1 = (float)ptr[0], x2 = (float)ptr[32];
  ptr[0]  = (_Float16)(x1 * cv - x2 * sv);
  ptr[32] = (_Float16)(x2 * cv + x1 * sv);
}

// ================= flash attention (scale=1.0, no mask) =================
// q: [RS, QD] f16, k: [RS, KD] f16, vT: [KD, RS] f16, o: [RS, QD] f16
// grid(S/64, Hq, B), block 128 (4 waves, one head, 64 q rows).
// 64-key tiles staged once per block into LDS (async -> ds reads).
__global__ void __launch_bounds__(128) attn_kernel(const _Float16* __restrict__ q,
                                                   const _Float16* __restrict__ k,
                                                   const _Float16* __restrict__ vT,
                                                   _Float16* __restrict__ o) {
  __shared__ _Float16 kTile[64 * 64];      // [key][d]   8 KB
  __shared__ _Float16 vTile[64 * 64];      // [d][key]   8 KB (transposed)
  __shared__ _Float16 pTile[4][16 * 64];   // wave-private P tiles, 8 KB

  int tid  = threadIdx.x;
  int lane = tid & 31;
  int wave = tid >> 5;
  int hq   = blockIdx.y;
  int b    = blockIdx.z;
  int hk   = hq >> 2;  // G = 4
  int q0   = blockIdx.x * 64 + wave * 16;
  int qrow = b * S + q0;

  v16h aq0 = load_a_frag(q, QD, qrow, hq * 64 + 0, lane);
  v16h aq1 = load_a_frag(q, QD, qrow, hq * 64 + 32, lane);

  v8f acc[4] = {};
  float mrun[8], lrun[8];
#pragma unroll
  for (int r = 0; r < 8; ++r) { mrun[r] = -1e30f; lrun[r] = 0.0f; }

  _Float16* pT = pTile[wave];
  int n  = lane & 15;
  int mb = (lane >> 4) * 8;
  int kb = (lane >> 4) * 16;
  int m  = lane & 15, hh = lane >> 4;

  for (int kt = 0; kt < S; kt += 64) {
    int krow = b * S + kt;

    // ---- cooperative staging: 512 x b128 chunks (4 per thread per tile) ----
#pragma unroll
    for (int c = 0; c < 4; ++c) {
      int chunk = c * 128 + tid;        // 0..511
      int row   = chunk >> 3;           // 0..63
      int off   = (chunk & 7) * 8;      // half offset within 128B row
      stage_b128(&kTile[row * 64 + off],
                 k + (size_t)(krow + row) * KD + hk * 64 + off);
      stage_b128(&vTile[row * 64 + off],
                 vT + (size_t)(hk * 64 + row) * RS + krow + off);
    }
    stage_wait_barrier();

    // ---- scores: four 16x16 tiles over 64 keys (8 wmma) ----
    v8f sc[4] = {};
#pragma unroll
    for (int t = 0; t < 4; ++t) {
      v16h bk0 = *(const v16h*)&kTile[(t * 16 + n) * 64 + 0 + kb];
      v16h bk1 = *(const v16h*)&kTile[(t * 16 + n) * 64 + 32 + kb];
      sc[t] = wmma16x16x32(aq0, bk0, sc[t]);
      sc[t] = wmma16x16x32(aq1, bk1, sc[t]);
    }

    // ---- online softmax: one reduction round per 64 keys ----
#pragma unroll
    for (int r = 0; r < 8; ++r) {
      float mx = fmaxf(fmaxf(sc[0][r], sc[1][r]), fmaxf(sc[2][r], sc[3][r]));
#pragma unroll
      for (int d = 1; d < 16; d <<= 1) mx = fmaxf(mx, __shfl_xor(mx, d, 32));
      float mnew = fmaxf(mrun[r], mx);
      float sum = 0.0f;
#pragma unroll
      for (int t = 0; t < 4; ++t) {
        float p = __expf(sc[t][r] - mnew);
        sc[t][r] = p;
        sum += p;
      }
#pragma unroll
      for (int d = 1; d < 16; d <<= 1) sum += __shfl_xor(sum, d, 32);
      float scale = __expf(mrun[r] - mnew);
      mrun[r] = mnew;
      lrun[r] = lrun[r] * scale + sum;
#pragma unroll
      for (int j = 0; j < 4; ++j) acc[j][r] *= scale;
    }

    // ---- relayout P: C-layout -> LDS -> two A-fragments (wave-private) ----
#pragma unroll
    for (int r = 0; r < 8; ++r) {
#pragma unroll
      for (int t = 0; t < 4; ++t)
        pT[(mb + r) * 64 + t * 16 + n] = (_Float16)sc[t][r];
    }
    HFrag pf[2];
#pragma unroll
    for (int kc = 0; kc < 2; ++kc) {
      pf[kc].h[0] = *(const v8h*)&pT[m * 64 + kc * 32 + hh * 16];
      pf[kc].h[1] = *(const v8h*)&pT[m * 64 + kc * 32 + hh * 16 + 8];
    }

    // ---- out += P(16x64) @ V(64x64)  (8 wmma) ----
#pragma unroll
    for (int kc = 0; kc < 2; ++kc) {
#pragma unroll
      for (int j = 0; j < 4; ++j) {
        v16h bv = *(const v16h*)&vTile[(j * 16 + n) * 64 + kc * 32 + kb];
        acc[j] = wmma16x16x32(pf[kc].v, bv, acc[j]);
      }
    }

    __syncthreads();  // tiles consumed; safe to restage
  }

  // ---- epilogue: normalize and store f16 ----
#pragma unroll
  for (int j = 0; j < 4; ++j) {
#pragma unroll
    for (int r = 0; r < 8; ++r) {
      float oval = acc[j][r] / lrun[r];
      o[(size_t)(qrow + mb + r) * QD + hq * 64 + j * 16 + n] = (_Float16)oval;
    }
  }
}

// ============================== launcher ==============================
extern "C" void kernel_launch(void* const* d_in, const int* in_sizes, int n_in,
                              void* d_out, int out_size, void* d_ws, size_t ws_size,
                              hipStream_t stream) {
  const float* x   = (const float*)d_in[0];
  const int*   qp  = (const int*)d_in[1];
  const float* Wq  = (const float*)d_in[2];
  const float* Wk  = (const float*)d_in[3];
  const float* Wv  = (const float*)d_in[4];
  const float* Wo  = (const float*)d_in[5];
  float*       out = (float*)d_out;

  // ---- workspace carve-up (f16) ----
  _Float16* xh  = (_Float16*)d_ws;          // RS*E
  _Float16* wqT = xh  + (size_t)RS * E;     // QD*E
  _Float16* wkT = wqT + (size_t)QD * E;     // KD*E
  _Float16* wvT = wkT + (size_t)KD * E;     // KD*E
  _Float16* woT = wvT + (size_t)KD * E;     // E*QD
  _Float16* qb  = woT + (size_t)E * QD;     // RS*QD
  _Float16* kbf = qb  + (size_t)RS * QD;    // RS*KD
  _Float16* vT  = kbf + (size_t)RS * KD;    // KD*RS
  _Float16* ab  = vT  + (size_t)KD * RS;    // RS*QD

  // ---- stage operands in f16 (x as-is, weights transposed) ----
  {
    int n4 = (RS * E) / 4;
    cvt_f32_f16<<<(n4 + 255) / 256, 256, 0, stream>>>(x, xh, n4);
    int nq = E * QD, nk = E * KD;
    transpose_cvt<<<(nq + 255) / 256, 256, 0, stream>>>(Wq, wqT, E, QD);
    transpose_cvt<<<(nk + 255) / 256, 256, 0, stream>>>(Wk, wkT, E, KD);
    transpose_cvt<<<(nk + 255) / 256, 256, 0, stream>>>(Wv, wvT, E, KD);
    transpose_cvt<<<(nq + 255) / 256, 256, 0, stream>>>(Wo, woT, QD, E);
  }

  // ---- projections (V written transposed for the PV WMMA layout) ----
  gemm_tn<_Float16, false><<<dim3(QD / 256, RS / 32), 256, 0, stream>>>(xh, wqT, qb, RS, QD, E);
  gemm_tn<_Float16, false><<<dim3(KD / 256, RS / 32), 256, 0, stream>>>(xh, wkT, kbf, RS, KD, E);
  gemm_tn<_Float16, true ><<<dim3(KD / 256, RS / 32), 256, 0, stream>>>(xh, wvT, vT, RS, KD, E);

  // ---- RoPE on q and k ----
  int tq = RS * Hq * 32;
  rope_f16<<<(tq + 255) / 256, 256, 0, stream>>>(qb, qp, Hq, tq);
  int tk = RS * Hk * 32;
  rope_f16<<<(tk + 255) / 256, 256, 0, stream>>>(kbf, qp, Hk, tk);

  // ---- attention ----
  attn_kernel<<<dim3(S / 64, Hq, Bn), dim3(128), 0, stream>>>(qb, kbf, vT, ab);

  // ---- output projection (fp32 out) ----
  gemm_tn<float, false><<<dim3(E / 256, RS / 32), 256, 0, stream>>>(ab, woT, out, RS, E, QD);
}
